// UltraTransformerBlock_55508157333992
// MI455X (gfx1250) — matrix operand order repaired and checked by reading, not compiled, not run on hardware
//
#include <hip/hip_runtime.h>
#include <math.h>

typedef _Float16 f16;
typedef __attribute__((ext_vector_type(16))) _Float16 v16h;
typedef __attribute__((ext_vector_type(8)))  _Float16 v8h;
typedef __attribute__((ext_vector_type(8)))  float    v8f;

union V16 { v16h v; v8h h[2]; };

#define D_MODEL 768
#define NH      12
#define NKV     4
#define HD      64
#define FF_DIM  2048
#define NREP    3
#define SEQ     2048

// ---------------------------------------------------------------------------
// fp32 -> f16 conversion (weights)
// ---------------------------------------------------------------------------
__global__ void f32_to_f16(const float* __restrict__ in, f16* __restrict__ out, int n) {
    int i = blockIdx.x * blockDim.x + threadIdx.x;
    if (i < n) out[i] = (f16)in[i];
}

// ---------------------------------------------------------------------------
// RMSNorm (reference semantics: norm = sqrt(sum x^2)*n^-1/2; y = g*x/(norm+eps))
// one block per row of 768, writes f16
// ---------------------------------------------------------------------------
__global__ __launch_bounds__(256) void rmsnorm_f16(const float* __restrict__ x,
                                                   const float* __restrict__ g,
                                                   f16* __restrict__ out, int M) {
    __shared__ float red[256];
    int row = blockIdx.x;
    int t = threadIdx.x;
    const float* xr = x + (size_t)row * D_MODEL;
    float s = 0.0f;
    for (int i = t; i < D_MODEL; i += 256) { float v = xr[i]; s += v * v; }
    red[t] = s;
    __syncthreads();
    for (int o = 128; o > 0; o >>= 1) {
        if (t < o) red[t] += red[t + o];
        __syncthreads();
    }
    float norm = sqrtf(red[0]) * 0.03608439182435161f;   // * 768^-0.5
    float inv  = 1.0f / (norm + 1e-6f);
    for (int i = t; i < D_MODEL; i += 256)
        out[(size_t)row * D_MODEL + i] = (f16)(g[i] * xr[i] * inv);
}

// ---------------------------------------------------------------------------
// WMMA GEMM: Y[M,N] = A[M,K] @ W[N,K]^T (+ residual[M,N])
// block = 128 threads (4 waves); each wave: 16 rows x 64 cols; K-step 32.
// B tile (64 cols x 32 k = 4KB) is staged in LDS with the CDNA5 async
// load-to-LDS path (ASYNCcnt), double-buffered so DMA overlaps WMMA.
// ---------------------------------------------------------------------------
template <bool OUT_F16>
__global__ __launch_bounds__(128) void gemm_xwT(const f16* __restrict__ A,
                                                const f16* __restrict__ W,
                                                float* __restrict__ outF,
                                                f16* __restrict__ outH,
                                                const float* __restrict__ residual,
                                                int M, int N, int K) {
    __shared__ __align__(32) f16 sB[2][64 * 32];

    int tid  = threadIdx.x;
    int wave = tid >> 5;
    int L    = tid & 31;
    int l    = L & 15;
    int hi   = L >> 4;
    int m0   = blockIdx.y * 64 + wave * 16;
    int n0   = blockIdx.x * 64;

    const f16* arow = A + (size_t)(m0 + l) * K;

    // Async-fill sB[buf] with W[n0..n0+63][k0..k0+31]; 256 x 16B chunks,
    // 2 chunks per thread -> 2 async b128 ops per wave (ASYNCcnt += 2).
    auto fill = [&](int buf, int k0) {
#pragma unroll
        for (int ii = 0; ii < 2; ii++) {
            int ch  = tid + ii * 128;
            int row = ch >> 2;
            int c   = (ch & 3) * 8;
            const f16* gsrc = W + (size_t)(n0 + row) * K + k0 + c;
            f16* ldst = &sB[buf][row * 32 + c];
            unsigned long long ga = (unsigned long long)(uintptr_t)gsrc;
            unsigned la = (unsigned)(uintptr_t)ldst;   // flat->LDS: addr[31:0]
            asm volatile("global_load_async_to_lds_b128 %0, %1, off"
                         :: "v"(la), "v"(ga) : "memory");
        }
    };

    fill(0, 0);

    v8f acc[4] = {};
    for (int k0 = 0; k0 < K; k0 += 32) {
        int buf = (k0 >> 5) & 1;
        if (k0 + 32 < K) {
            fill(buf ^ 1, k0 + 32);
            // 2 newest (next tile) may stay in flight; current tile complete.
            asm volatile("s_wait_asynccnt 0x2" ::: "memory");
            __builtin_prefetch(arow + k0 + 32, 0, 1);   // global_prefetch_b8
        } else {
            asm volatile("s_wait_asynccnt 0x0" ::: "memory");
        }
        __syncthreads();

        // A frag: lane row m0+l, element e -> k0+16*hi+e
        V16 a;
        a.h[0] = *(const v8h*)(arow + k0 + 16 * hi);
        a.h[1] = *(const v8h*)(arow + k0 + 16 * hi + 8);
#pragma unroll
        for (int t = 0; t < 4; t++) {
            // B frag from LDS: lane col n0+16t+l, element e -> k = k0 + e + 16*hi
            v16h b = *(const v16h*)(&sB[buf][(16 * t + l) * 32 + 16 * hi]);
            acc[t] = __builtin_amdgcn_wmma_f32_16x16x32_f16(
                false, a.v, false, b, (short)0, acc[t], false, false);
        }
        __syncthreads();   // all waves done with sB[buf] before its refill
    }
#pragma unroll
    for (int t = 0; t < 4; t++) {
        int col = n0 + 16 * t + l;
#pragma unroll
        for (int r = 0; r < 8; r++) {
            int row = m0 + r + 8 * hi;      // C/D layout: VGPR r = row r (+8 for hi lanes)
            float v = acc[t][r];
            if (residual) v += residual[(size_t)row * N + col];
            if constexpr (OUT_F16) outH[(size_t)row * N + col] = (f16)v;
            else                   outF[(size_t)row * N + col] = v;
        }
    }
}

// ---------------------------------------------------------------------------
// RoPE, in place on f16 [M, nheads*64]; position = row % SEQ
// ---------------------------------------------------------------------------
__global__ void rope_inplace(f16* __restrict__ p, int nheads, int stride, int M) {
    int idx = blockIdx.x * blockDim.x + threadIdx.x;
    int total = M * nheads * 32;
    if (idx >= total) return;
    int i    = idx & 31;
    int head = (idx >> 5) % nheads;
    int row  = idx / (32 * nheads);
    int s    = row & (SEQ - 1);
    float invf = powf(10000.0f, -(float)i * (1.0f / 32.0f));
    float ang  = (float)s * invf;
    float c = cosf(ang), sn = sinf(ang);
    f16* base = p + (size_t)row * stride + head * HD;
    float q0 = (float)base[i];
    float q1 = (float)base[i + 32];
    base[i]      = (f16)(q0 * c - q1 * sn);
    base[i + 32] = (f16)(q1 * c + q0 * sn);
}

// ---------------------------------------------------------------------------
// Causal flash attention, GQA (12 q heads over 4 kv heads).
// One wave per (16-query tile, head). Key blocks of 32.
// ---------------------------------------------------------------------------
__global__ __launch_bounds__(32) void flash_attn(const f16* __restrict__ q,
                                                 const f16* __restrict__ k,
                                                 const f16* __restrict__ v,
                                                 f16* __restrict__ o,
                                                 int S, int qtPerB) {
    __shared__ __align__(32) float sS[16 * 32];
    __shared__ __align__(32) f16   sP[16 * 32];
    __shared__ __align__(32) f16   sVt[64 * 32];   // V^T : [d][key]
    __shared__ float sM[16], sLs[16], sC[16];

    int L  = threadIdx.x;
    int l  = L & 15;
    int hi = L >> 4;
    int qt = blockIdx.x;
    int h  = blockIdx.y;
    int b  = qt / qtPerB;
    int s0 = (qt % qtPerB) * 16;
    int hk = h / NREP;

    // Q tile as two A-frags (d 0..31 and 32..63), kept live across the key loop
    const f16* qrow = q + (size_t)(b * S + s0 + l) * (NH * HD) + h * HD;
    V16 a0, a1;
    a0.h[0] = *(const v8h*)(qrow + 16 * hi);
    a0.h[1] = *(const v8h*)(qrow + 16 * hi + 8);
    a1.h[0] = *(const v8h*)(qrow + 32 + 16 * hi);
    a1.h[1] = *(const v8h*)(qrow + 40 + 16 * hi);

    v8f oacc[4] = {};
    if (L < 16) { sM[L] = -3.0e38f; sLs[L] = 0.0f; }
    __syncthreads();

    for (int j0 = 0; j0 <= s0 + 15; j0 += 32) {
        // ---- stage V^T into LDS (lane L owns key j0+L; coalesced global reads).
        // Load all 8 chunks into registers first so loads overlap, then scatter.
        {
            int j  = j0 + L;
            int jc = j < S ? j : S - 1;               // clamped rows are masked later
            const f16* vrow = v + (size_t)(b * S + jc) * (NKV * HD) + hk * HD;
            v8h tv[8];
#pragma unroll
            for (int dc = 0; dc < 8; dc++) tv[dc] = *(const v8h*)(vrow + dc * 8);
#pragma unroll
            for (int dc = 0; dc < 8; dc++)
#pragma unroll
                for (int e = 0; e < 8; e++) sVt[(dc * 8 + e) * 32 + L] = tv[dc][e];
        }
        // ---- S = Q K^T : two 16x16 tiles, 2 WMMAs each (HD=64 => K-steps 0..31, 32..63)
#pragma unroll
        for (int t = 0; t < 2; t++) {
            int jc = j0 + 16 * t + l;
            if (jc >= S) jc = S - 1;
            const f16* krow = k + (size_t)(b * S + jc) * (NKV * HD) + hk * HD;
            v16h b0 = *(const v16h*)(krow + 16 * hi);        // d = 16*hi + e
            v16h b1 = *(const v16h*)(krow + 32 + 16 * hi);
            v8f st = {};
            st = __builtin_amdgcn_wmma_f32_16x16x32_f16(false, a0.v, false, b0, (short)0, st, false, false);
            st = __builtin_amdgcn_wmma_f32_16x16x32_f16(false, a1.v, false, b1, (short)0, st, false, false);
#pragma unroll
            for (int r = 0; r < 8; r++) sS[(r + 8 * hi) * 32 + 16 * t + l] = st[r];
        }
        __syncthreads();
        // ---- online softmax: lane L<16 owns query row s0+L ----
        if (L < 16) {
            int   qg   = s0 + L;
            float mOld = sM[L];
            float scv[32];
            float mx = -3.0e38f;
#pragma unroll
            for (int c = 0; c < 32; c++) {
                int key = j0 + c;
                float sc = (key <= qg) ? sS[L * 32 + c] * 0.125f : -3.0e38f;
                scv[c] = sc;
                mx = fmaxf(mx, sc);
            }
            float mNew = fmaxf(mOld, mx);
            float corr = __expf(mOld - mNew);
            float sum = 0.0f;
#pragma unroll
            for (int c = 0; c < 32; c++) {
                float pv = (scv[c] > -1.0e38f) ? __expf(scv[c] - mNew) : 0.0f;
                sP[L * 32 + c] = (f16)pv;
                sum += pv;
            }
            sLs[L] = sLs[L] * corr + sum;
            sM[L]  = mNew;
            sC[L]  = corr;
        }
        __syncthreads();
        // ---- rescale running O, then O += P V ----
        float cr[8];
#pragma unroll
        for (int r = 0; r < 8; r++) cr[r] = sC[r + 8 * hi];
#pragma unroll
        for (int t = 0; t < 4; t++)
#pragma unroll
            for (int r = 0; r < 8; r++) oacc[t][r] *= cr[r];

        V16 ap;  // P as A-frag: row l, element e -> key 16*hi+e
        ap.h[0] = *(const v8h*)(sP + l * 32 + 16 * hi);
        ap.h[1] = *(const v8h*)(sP + l * 32 + 16 * hi + 8);
#pragma unroll
        for (int t = 0; t < 4; t++) {
            // B = V tile: col d = 16t+l, element e -> key e+16*hi (contiguous in sVt)
            v16h bv = *(const v16h*)(sVt + (16 * t + l) * 32 + 16 * hi);
            oacc[t] = __builtin_amdgcn_wmma_f32_16x16x32_f16(
                false, ap.v, false, bv, (short)0, oacc[t], false, false);
        }
        __syncthreads();
    }
    // ---- finalize: O /= l, store f16 [B*S, NH*HD] ----
    float rl[8];
#pragma unroll
    for (int r = 0; r < 8; r++) rl[r] = 1.0f / sLs[r + 8 * hi];
#pragma unroll
    for (int t = 0; t < 4; t++)
#pragma unroll
        for (int r = 0; r < 8; r++) {
            int row = b * S + s0 + r + 8 * hi;
            o[(size_t)row * (NH * HD) + h * HD + 16 * t + l] = (f16)(oacc[t][r] * rl[r]);
        }
}

// ---------------------------------------------------------------------------
// h = silu(gate) * up   (f16 in/out)
// ---------------------------------------------------------------------------
__global__ void silu_mul(const f16* __restrict__ g, const f16* __restrict__ u,
                         f16* __restrict__ o, int n) {
    int i = blockIdx.x * blockDim.x + threadIdx.x;
    if (i >= n) return;
    float gv = (float)g[i];
    float s  = gv / (1.0f + __expf(-gv));
    o[i] = (f16)(s * (float)u[i]);
}

// ---------------------------------------------------------------------------
extern "C" void kernel_launch(void* const* d_in, const int* in_sizes, int n_in,
                              void* d_out, int out_size, void* d_ws, size_t ws_size,
                              hipStream_t stream) {
    const float* x     = (const float*)d_in[0];
    const float* Wq    = (const float*)d_in[1];
    const float* Wk    = (const float*)d_in[2];
    const float* Wv    = (const float*)d_in[3];
    const float* Wo    = (const float*)d_in[4];
    const float* Wgate = (const float*)d_in[5];
    const float* Wup   = (const float*)d_in[6];
    const float* Wdown = (const float*)d_in[7];
    const float* g1    = (const float*)d_in[8];
    const float* g2    = (const float*)d_in[9];
    float* out = (float*)d_out;

    const int M = in_sizes[0] / D_MODEL;   // B*S = 4096
    const int S = SEQ;
    const int qtPerB = S / 16;

    // ---- bump allocator over d_ws (256B aligned) ----
    char* p = (char*)d_ws;
    auto alloc = [&](size_t bytes) -> void* {
        void* r = (void*)p;
        p += (bytes + 255) & ~(size_t)255;
        return r;
    };
    f16* Wq_h   = (f16*)alloc((size_t)768 * 768 * 2);
    f16* Wk_h   = (f16*)alloc((size_t)256 * 768 * 2);
    f16* Wv_h   = (f16*)alloc((size_t)256 * 768 * 2);
    f16* Wo_h   = (f16*)alloc((size_t)768 * 768 * 2);
    f16* Wg_h   = (f16*)alloc((size_t)FF_DIM * 768 * 2);
    f16* Wu_h   = (f16*)alloc((size_t)FF_DIM * 768 * 2);
    f16* Wd_h   = (f16*)alloc((size_t)768 * FF_DIM * 2);
    f16* xn_h   = (f16*)alloc((size_t)M * 768 * 2);
    f16* q_h    = (f16*)alloc((size_t)M * 768 * 2);
    f16* k_h    = (f16*)alloc((size_t)M * 256 * 2);
    f16* v_h    = (f16*)alloc((size_t)M * 256 * 2);
    f16* attn_h = (f16*)alloc((size_t)M * 768 * 2);
    float* x1   = (float*)alloc((size_t)M * 768 * 4);
    f16* xn2_h  = (f16*)alloc((size_t)M * 768 * 2);
    f16* gate_h = (f16*)alloc((size_t)M * FF_DIM * 2);
    f16* up_h   = (f16*)alloc((size_t)M * FF_DIM * 2);
    f16* hff_h  = (f16*)alloc((size_t)M * FF_DIM * 2);

    auto cvt = [&](const float* src, f16* dst, int n) {
        f32_to_f16<<<(n + 255) / 256, 256, 0, stream>>>(src, dst, n);
    };
    cvt(Wq, Wq_h, 768 * 768);
    cvt(Wk, Wk_h, 256 * 768);
    cvt(Wv, Wv_h, 256 * 768);
    cvt(Wo, Wo_h, 768 * 768);
    cvt(Wgate, Wg_h, FF_DIM * 768);
    cvt(Wup, Wu_h, FF_DIM * 768);
    cvt(Wdown, Wd_h, 768 * FF_DIM);

    // ---- attention sublayer ----
    rmsnorm_f16<<<M, 256, 0, stream>>>(x, g1, xn_h, M);

    gemm_xwT<true><<<dim3(768 / 64, M / 64), 128, 0, stream>>>(xn_h, Wq_h, nullptr, q_h, nullptr, M, 768, 768);
    gemm_xwT<true><<<dim3(256 / 64, M / 64), 128, 0, stream>>>(xn_h, Wk_h, nullptr, k_h, nullptr, M, 256, 768);
    gemm_xwT<true><<<dim3(256 / 64, M / 64), 128, 0, stream>>>(xn_h, Wv_h, nullptr, v_h, nullptr, M, 256, 768);

    {
        int tq = M * NH * 32;
        rope_inplace<<<(tq + 255) / 256, 256, 0, stream>>>(q_h, NH, 768, M);
        int tk = M * NKV * 32;
        rope_inplace<<<(tk + 255) / 256, 256, 0, stream>>>(k_h, NKV, 256, M);
    }

    flash_attn<<<dim3(M / 16, NH), 32, 0, stream>>>(q_h, k_h, v_h, attn_h, S, qtPerB);

    // x1 = x + attn @ Wo^T
    gemm_xwT<false><<<dim3(768 / 64, M / 64), 128, 0, stream>>>(attn_h, Wo_h, x1, nullptr, x, M, 768, 768);

    // ---- FFN sublayer ----
    rmsnorm_f16<<<M, 256, 0, stream>>>(x1, g2, xn2_h, M);

    gemm_xwT<true><<<dim3(FF_DIM / 64, M / 64), 128, 0, stream>>>(xn2_h, Wg_h, nullptr, gate_h, nullptr, M, FF_DIM, 768);
    gemm_xwT<true><<<dim3(FF_DIM / 64, M / 64), 128, 0, stream>>>(xn2_h, Wu_h, nullptr, up_h, nullptr, M, FF_DIM, 768);

    silu_mul<<<(M * FF_DIM + 255) / 256, 256, 0, stream>>>(gate_h, up_h, hff_h, M * FF_DIM);

    // out = x1 + h @ Wdown^T
    gemm_xwT<false><<<dim3(768 / 64, M / 64), 128, 0, stream>>>(hff_h, Wd_h, out, nullptr, x1, M, 768, FF_DIM);

    (void)n_in; (void)out_size; (void)ws_size;
}
